// EncoderBlock_5488968204521
// MI455X (gfx1250) — matrix operand order repaired and checked by reading, not compiled, not run on hardware
//
#include <hip/hip_runtime.h>
#include <hip/hip_bf16.h>

// ---------------------------------------------------------------------------
// Types
// ---------------------------------------------------------------------------
typedef _Float16 half_t;
typedef __attribute__((ext_vector_type(8)))  _Float16 v8h;
typedef __attribute__((ext_vector_type(16))) _Float16 v16h;
typedef __attribute__((ext_vector_type(8)))  float    v8f;
typedef int v4i __attribute__((vector_size(16)));

__device__ __forceinline__ v16h cat16(v8h lo, v8h hi) {
  return __builtin_shufflevector(lo, hi, 0,1,2,3,4,5,6,7,8,9,10,11,12,13,14,15);
}

// CDNA5 async global->LDS copy (ASYNCcnt-tracked). Guarded so either toolchain
// compiles: falls back to a synchronous VGPR round-trip if the builtin is absent.
// Probe-discovered signature: (v4i AS1* src, v4i AS3* dst, imm offset, imm cpol).
#if defined(__AMDGCN__) && __has_builtin(__builtin_amdgcn_global_load_async_to_lds_b128)
#define USE_ASYNC_LDS 1
#else
#define USE_ASYNC_LDS 0
#endif

__device__ __forceinline__ void copy16(const half_t* __restrict__ g, half_t* l) {
#if USE_ASYNC_LDS
  __builtin_amdgcn_global_load_async_to_lds_b128(
      (__attribute__((address_space(1))) v4i*)g,
      (__attribute__((address_space(3))) v4i*)l, 0, 0);
#else
  *(v8h*)l = *(const v8h*)g;
#endif
}

__device__ __forceinline__ void wait_async0() {
#if USE_ASYNC_LDS
#if __has_builtin(__builtin_amdgcn_s_wait_asynccnt)
  __builtin_amdgcn_s_wait_asynccnt(0);
#else
  asm volatile("s_wait_asynccnt 0" ::: "memory");
#endif
#endif
}

// Problem constants
constexpr int kS   = 2048;      // sequence
constexpr int kF   = 1024;      // model dim
constexpr int kHD  = 32;        // head dim
constexpr int kNH  = 32;        // heads
constexpr int kHID = 4096;      // ffn hidden
constexpr int kTOK = 4096;      // B*S tokens

// ---------------------------------------------------------------------------
// f32 -> f16 conversion (8 elements / thread, vectorized)
// ---------------------------------------------------------------------------
__global__ void __launch_bounds__(256)
cvt_f32_f16(const float* __restrict__ in, half_t* __restrict__ out, int n8) {
  int i = blockIdx.x * blockDim.x + threadIdx.x;
  if (i >= n8) return;
  const float4* p = (const float4*)in + (size_t)i * 2;
  float4 a = p[0], b = p[1];
  v8h o;
  o[0] = (half_t)a.x; o[1] = (half_t)a.y; o[2] = (half_t)a.z; o[3] = (half_t)a.w;
  o[4] = (half_t)b.x; o[5] = (half_t)b.y; o[6] = (half_t)b.z; o[7] = (half_t)b.w;
  *((v8h*)out + i) = o;
}

// ---------------------------------------------------------------------------
// Tiled WMMA GEMM:  C[M,N] = A[M,K] (row-major, f16) * Bw[N,K]^T + bias[N]
// Block tile 128x128, BK=64 (two WMMA k-steps per staged tile).
// 8 waves, each wave: 64(M) x 32(N) = 4x2 WMMA tiles -> 16 WMMAs per barrier.
// Double-buffered LDS with async global->LDS staging.
// ---------------------------------------------------------------------------
constexpr int BM = 128, BN = 128, BK = 64;
constexpr int LDSK = 72;   // padded halfs per row (144B, keeps 16B alignment)

template <bool STORE_HALF, bool RELU>
__global__ void __launch_bounds__(256)
gemm_wmma(const half_t* __restrict__ A, const half_t* __restrict__ Bw,
          const float* __restrict__ bias, void* __restrict__ Cout,
          int M, int N, int K) {
  __shared__ half_t As[2][BM * LDSK];
  __shared__ half_t Bs[2][BN * LDSK];

  const int tid  = threadIdx.x;
  const int lane = tid & 31;
  const int wid  = tid >> 5;
  const int wm   = wid >> 2;          // 0..1  -> 64 rows each
  const int wn   = wid & 3;           // 0..3  -> 32 cols each
  const int l15  = lane & 15;
  const int lh   = lane >> 4;         // 0/1 half of wave
  const int bm0  = blockIdx.y * BM;
  const int bn0  = blockIdx.x * BN;

  v8f zv = {0.f,0.f,0.f,0.f,0.f,0.f,0.f,0.f};
  v8f acc[4][2];
#pragma unroll
  for (int i = 0; i < 4; ++i)
#pragma unroll
    for (int j = 0; j < 2; ++j) acc[i][j] = zv;

  // Stage one 128x64 tile of A and B into LDS buffer `buf`.
  auto stage = [&](int k0, int buf) {
#pragma unroll
    for (int i = 0; i < 4; ++i) {
      int c  = tid + i * 256;          // 1024 chunks of 8 halfs
      int r  = c >> 3;                 // 0..127
      int cc = (c & 7) * 8;            // 0..56
      copy16(&A [(size_t)(bm0 + r) * K + k0 + cc], &As[buf][r * LDSK + cc]);
      copy16(&Bw[(size_t)(bn0 + r) * K + k0 + cc], &Bs[buf][r * LDSK + cc]);
    }
  };

  const int nk = K / BK;
  stage(0, 0);
  wait_async0();
  __syncthreads();

  for (int ki = 0; ki < nk; ++ki) {
    const half_t* Ac = As[ki & 1];
    const half_t* Bc = Bs[ki & 1];
    if (ki + 1 < nk) stage((ki + 1) * BK, (ki + 1) & 1);

#pragma unroll
    for (int ks = 0; ks < BK; ks += 32) {
      // A fragments: lane -> row = base+(l&15); K elems {kb..kb+7, kb+16..kb+23}
      v16h af[4];
#pragma unroll
      for (int t = 0; t < 4; ++t) {
        int row = wm * 64 + t * 16 + l15;
        int kb  = ks + lh * 8;
        af[t] = cat16(*(const v8h*)&Ac[row * LDSK + kb],
                      *(const v8h*)&Ac[row * LDSK + kb + 16]);
      }
      // B fragments: lane -> col = base+(l&15); K elems {kd..kd+7, kd+16..kd+23}
      v16h bf[2];
#pragma unroll
      for (int t = 0; t < 2; ++t) {
        int row = wn * 32 + t * 16 + l15;
        int kd  = ks + lh * 8;
        bf[t] = cat16(*(const v8h*)&Bc[row * LDSK + kd],
                      *(const v8h*)&Bc[row * LDSK + kd + 16]);
      }
#pragma unroll
      for (int i = 0; i < 4; ++i)
#pragma unroll
        for (int j = 0; j < 2; ++j)
          acc[i][j] = __builtin_amdgcn_wmma_f32_16x16x32_f16(
              false, af[i], false, bf[j], (short)0, acc[i][j], false, false);
    }

    wait_async0();
    __syncthreads();
  }

  // Epilogue: C layout -> lane (l&15)=col, vgpr r -> row r + lh*8
#pragma unroll
  for (int i = 0; i < 4; ++i)
#pragma unroll
    for (int j = 0; j < 2; ++j) {
      int col = bn0 + wn * 32 + j * 16 + l15;
      float bv = bias[col];
#pragma unroll
      for (int r = 0; r < 8; ++r) {
        int row = bm0 + wm * 64 + i * 16 + r + lh * 8;
        float v = acc[i][j][r] + bv;
        if (RELU) v = fmaxf(v, 0.0f);
        if (STORE_HALF) ((half_t*)Cout)[(size_t)row * N + col] = (half_t)v;
        else            ((float*) Cout)[(size_t)row * N + col] = v;
      }
    }
}

// ---------------------------------------------------------------------------
// Flash attention: 32 heads, head_dim 32, scale 1/sqrt(32).
// Block = (128 query rows) x one (batch,head). 8 waves x 16 rows.
// Streams 32 keys per step with online softmax; WMMA for QK^T and P*V.
// ---------------------------------------------------------------------------
constexpr int ALD = 40;   // LDS row pad for attention tiles (80B)

__global__ void __launch_bounds__(256)
attn_flash(const half_t* __restrict__ Q, const half_t* __restrict__ K,
           const half_t* __restrict__ V, half_t* __restrict__ O) {
  __shared__ half_t Ks[32 * ALD];         // keys x dims
  __shared__ half_t Vt[kHD * ALD];        // dims x keys (transposed)
  __shared__ half_t Ps[128 * ALD];        // per-wave 16-row P scratch

  const int tid = threadIdx.x, lane = tid & 31, wid = tid >> 5;
  const int l15 = lane & 15, lh = lane >> 4;
  const int bh = blockIdx.y;
  const int b  = bh >> 5;
  const int h  = bh & 31;
  const int q0 = blockIdx.x * 128;
  const size_t base = (size_t)b * kS * kF + (size_t)h * kHD;

  // Q fragment (A operand: 16 rows x 32 dims), loaded once from global
  const int kb = lh * 8;
  const int qrow = q0 + wid * 16 + l15;
  const half_t* qp = Q + base + (size_t)qrow * kF + kb;
  v16h qfrag = cat16(*(const v8h*)qp, *(const v8h*)(qp + 16));

  const float scale = 0.17677669529663687f;  // 1/sqrt(32)
  float m[8], l[8];
  v8f zv = {0.f,0.f,0.f,0.f,0.f,0.f,0.f,0.f};
  v8f oacc[2] = {zv, zv};
#pragma unroll
  for (int r = 0; r < 8; ++r) { m[r] = -1e30f; l[r] = 0.f; }

  for (int k0 = 0; k0 < kS; k0 += 32) {
    __syncthreads();
    if (tid < 128) {                     // stage K (async) and V (transposed)
      int key = tid >> 2, cc = (tid & 3) * 8;
      copy16(K + base + (size_t)(k0 + key) * kF + cc, &Ks[key * ALD + cc]);
      v8h vv = *(const v8h*)(V + base + (size_t)(k0 + key) * kF + cc);
#pragma unroll
      for (int j = 0; j < 8; ++j) Vt[(cc + j) * ALD + key] = vv[j];
    }
    wait_async0();
    __syncthreads();

    // scores = Q * K^T  (B operand: lane = key column, dims {kd..kd+7, kd+16..kd+23})
    v16h bk[2];
#pragma unroll
    for (int t = 0; t < 2; ++t) {
      int key = t * 16 + l15;
      int kd  = lh * 8;
      bk[t] = cat16(*(const v8h*)&Ks[key * ALD + kd],
                    *(const v8h*)&Ks[key * ALD + kd + 16]);
    }
    v8f s0 = __builtin_amdgcn_wmma_f32_16x16x32_f16(false, qfrag, false, bk[0],
                                                    (short)0, zv, false, false);
    v8f s1 = __builtin_amdgcn_wmma_f32_16x16x32_f16(false, qfrag, false, bk[1],
                                                    (short)0, zv, false, false);

    // Online softmax (row r lives in vgpr slot r across the 16 lanes of a half)
#pragma unroll
    for (int r = 0; r < 8; ++r) {
      float a0 = s0[r] * scale, a1 = s1[r] * scale;
      float mx = fmaxf(a0, a1);
#pragma unroll
      for (int off = 1; off < 16; off <<= 1) mx = fmaxf(mx, __shfl_xor(mx, off, 32));
      float mnew = fmaxf(m[r], mx);
      float corr = __expf(m[r] - mnew);
      float p0 = __expf(a0 - mnew), p1 = __expf(a1 - mnew);
      float ps = p0 + p1;
#pragma unroll
      for (int off = 1; off < 16; off <<= 1) ps += __shfl_xor(ps, off, 32);
      l[r] = l[r] * corr + ps;
      m[r] = mnew;
      oacc[0][r] *= corr;
      oacc[1][r] *= corr;
      int prow = wid * 16 + r + lh * 8;
      Ps[prow * ALD + l15]      = (half_t)p0;
      Ps[prow * ALD + 16 + l15] = (half_t)p1;
    }

    // P (A operand from wave-local LDS) times V (B operand from Vt)
    {
      int prow = wid * 16 + l15;
      v16h pfrag = cat16(*(const v8h*)&Ps[prow * ALD + kb],
                         *(const v8h*)&Ps[prow * ALD + kb + 16]);
#pragma unroll
      for (int t = 0; t < 2; ++t) {
        int dim = t * 16 + l15;
        int kk  = lh * 8;
        v16h vfrag = cat16(*(const v8h*)&Vt[dim * ALD + kk],
                           *(const v8h*)&Vt[dim * ALD + kk + 16]);
        oacc[t] = __builtin_amdgcn_wmma_f32_16x16x32_f16(
            false, pfrag, false, vfrag, (short)0, oacc[t], false, false);
      }
    }
  }

  // Normalize and write f16 output
#pragma unroll
  for (int t = 0; t < 2; ++t)
#pragma unroll
    for (int r = 0; r < 8; ++r) {
      int row = q0 + wid * 16 + r + lh * 8;
      int col = t * 16 + l15;
      float v = oacc[t][r] / l[r];
      O[base + (size_t)row * kF + col] = (half_t)v;
    }
}

// ---------------------------------------------------------------------------
// Residual + LayerNorm:  y = (x+res - mu)/(sigma + 1e-9)*alpha + beta
// One block (256 threads) per row of 1024.
// ---------------------------------------------------------------------------
template <bool EMIT_HALF>
__global__ void __launch_bounds__(256)
ln_residual(const float* __restrict__ xin, const float* __restrict__ res,
            const float* __restrict__ alpha, const float* __restrict__ beta,
            float* __restrict__ out, half_t* __restrict__ outh) {
  __shared__ float red1[8], red2[8];
  const int row = blockIdx.x, tid = threadIdx.x;
  const float* xr = xin + (size_t)row * kF;
  const float* rr = res + (size_t)row * kF;

  float vals[4];
  float sum = 0.f;
#pragma unroll
  for (int i = 0; i < 4; ++i) {
    int c = tid + i * 256;
    vals[i] = xr[c] + rr[c];
    sum += vals[i];
  }
#pragma unroll
  for (int off = 1; off < 32; off <<= 1) sum += __shfl_xor(sum, off, 32);
  if ((tid & 31) == 0) red1[tid >> 5] = sum;
  __syncthreads();
  float tot = 0.f;
#pragma unroll
  for (int w = 0; w < 8; ++w) tot += red1[w];
  float mu = tot * (1.0f / kF);

  float var = 0.f;
#pragma unroll
  for (int i = 0; i < 4; ++i) { float d = vals[i] - mu; var += d * d; }
#pragma unroll
  for (int off = 1; off < 32; off <<= 1) var += __shfl_xor(var, off, 32);
  if ((tid & 31) == 0) red2[tid >> 5] = var;
  __syncthreads();
  float vtot = 0.f;
#pragma unroll
  for (int w = 0; w < 8; ++w) vtot += red2[w];
  float sigma = sqrtf(vtot * (1.0f / kF));
  float inv = 1.0f / (sigma + 1e-9f);

#pragma unroll
  for (int i = 0; i < 4; ++i) {
    int c = tid + i * 256;
    float y = (vals[i] - mu) * inv * alpha[c] + beta[c];
    out[(size_t)row * kF + c] = y;
    if (EMIT_HALF) outh[(size_t)row * kF + c] = (half_t)y;
  }
}

// ---------------------------------------------------------------------------
// Launch
// ---------------------------------------------------------------------------
extern "C" void kernel_launch(void* const* d_in, const int* in_sizes, int n_in,
                              void* d_out, int out_size, void* d_ws, size_t ws_size,
                              hipStream_t stream) {
  (void)in_sizes; (void)n_in; (void)out_size; (void)ws_size;
  const float* x    = (const float*)d_in[0];
  const float* Wq   = (const float*)d_in[1];
  const float* bq   = (const float*)d_in[2];
  const float* Wk   = (const float*)d_in[3];
  const float* bk   = (const float*)d_in[4];
  const float* Wv   = (const float*)d_in[5];
  const float* bv   = (const float*)d_in[6];
  const float* Wo   = (const float*)d_in[7];
  const float* bo   = (const float*)d_in[8];
  const float* l0a  = (const float*)d_in[9];
  const float* l0b  = (const float*)d_in[10];
  const float* W0   = (const float*)d_in[11];
  const float* b0   = (const float*)d_in[12];
  const float* W1   = (const float*)d_in[13];
  const float* b1   = (const float*)d_in[14];
  const float* l1a  = (const float*)d_in[15];
  const float* l1b  = (const float*)d_in[16];

  char* ws = (char*)d_ws;
  const size_t MB = 1u << 20;
  half_t* xh   = (half_t*)(ws +   0 * MB);   // 8 MB
  half_t* wqh  = (half_t*)(ws +   8 * MB);   // 2 MB
  half_t* wkh  = (half_t*)(ws +  10 * MB);   // 2 MB
  half_t* wvh  = (half_t*)(ws +  12 * MB);   // 2 MB
  half_t* woh  = (half_t*)(ws +  14 * MB);   // 2 MB
  half_t* w0h  = (half_t*)(ws +  16 * MB);   // 8 MB
  half_t* w1h  = (half_t*)(ws +  24 * MB);   // 8 MB
  half_t* qh   = (half_t*)(ws +  32 * MB);   // 8 MB
  half_t* kh   = (half_t*)(ws +  40 * MB);   // 8 MB
  half_t* vh   = (half_t*)(ws +  48 * MB);   // 8 MB
  half_t* aoh  = (half_t*)(ws +  56 * MB);   // 8 MB
  float*  hpre = (float*) (ws +  64 * MB);   // 16 MB
  float*  h32  = (float*) (ws +  80 * MB);   // 16 MB
  half_t* h16  = (half_t*)(ws +  96 * MB);   // 8 MB
  half_t* f1h  = (half_t*)(ws + 104 * MB);   // 32 MB
  float*  fpre = (float*) (ws + 136 * MB);   // 16 MB

  // 1) f32 -> f16 casts
  auto cvt = [&](const float* src, half_t* dst, int n) {
    int n8 = n / 8;
    cvt_f32_f16<<<(n8 + 255) / 256, 256, 0, stream>>>(src, dst, n8);
  };
  cvt(x,  xh,  kTOK * kF);
  cvt(Wq, wqh, kF * kF);
  cvt(Wk, wkh, kF * kF);
  cvt(Wv, wvh, kF * kF);
  cvt(Wo, woh, kF * kF);
  cvt(W0, w0h, kHID * kF);
  cvt(W1, w1h, kF * kHID);

  // 2) QKV projections (f16 out with bias)
  dim3 gP(kF / BN, kTOK / BM);
  gemm_wmma<true, false><<<gP, 256, 0, stream>>>(xh, wqh, bq, qh, kTOK, kF, kF);
  gemm_wmma<true, false><<<gP, 256, 0, stream>>>(xh, wkh, bk, kh, kTOK, kF, kF);
  gemm_wmma<true, false><<<gP, 256, 0, stream>>>(xh, wvh, bv, vh, kTOK, kF, kF);

  // 3) Attention. NOTE: reference calls _mha(k, q, v) -> query is the
  //    k-projection, key is the q-projection.
  attn_flash<<<dim3(kS / 128, 2 * kNH), 256, 0, stream>>>(kh, qh, vh, aoh);

  // 4) Output projection (f32 out)
  gemm_wmma<false, false><<<gP, 256, 0, stream>>>(aoh, woh, bo, hpre, kTOK, kF, kF);

  // 5) LN0 with residual x -> h32 (f32) + h16 (f16)
  ln_residual<true><<<kTOK, 256, 0, stream>>>(hpre, x, l0a, l0b, h32, h16);

  // 6) FFN up (ReLU, f16 out)
  dim3 gU(kHID / BN, kTOK / BM);
  gemm_wmma<true, true><<<gU, 256, 0, stream>>>(h16, w0h, b0, f1h, kTOK, kHID, kF);

  // 7) FFN down (f32 out)
  gemm_wmma<false, false><<<gP, 256, 0, stream>>>(f1h, w1h, b1, fpre, kTOK, kF, kHID);

  // 8) LN1 with residual h32 -> d_out (f32)
  ln_residual<false><<<kTOK, 256, 0, stream>>>(fpre, h32, l1a, l1b, (float*)d_out, nullptr);
}
